// Network_29532195127320
// MI455X (gfx1250) — matrix operand order repaired and checked by reading, not compiled, not run on hardware
//
#include <hip/hip_runtime.h>
#include <hip/hip_bf16.h>
#include <math.h>

// ---------------------------------------------------------------------------
// MI455X (gfx1250) implementation.
//  - bf16 WMMA (16x16x32, f32 accum) for the two radial MLPs (~484 GFLOP).
//  - last r1 layer (500->2700) fused with einsum('euv,eu->ev') + scatter, so
//    the 1.1 GB "w" tensor is never materialized in HBM.
//  - double-buffered LDS staging, software pipelined; uses CDNA5
//    GLOBAL_LOAD_ASYNC_TO_LDS_B128 + s_wait_asynccnt when the builtin exists.
// ---------------------------------------------------------------------------

#define MULC    30
#define NBASIS  40
#define HDIM    500
#define HP      512      // padded hidden
#define K0P     64       // padded basis (40 -> 64)
#define NW1     2700
#define NW1P    2704     // padded to multiple of 16
#define NW1T    (NW1P/16)
#define NW2     60
#define NW2P    128
#define NGRAPH  64

typedef __attribute__((ext_vector_type(16))) __bf16       v16bf;
typedef __attribute__((ext_vector_type(8)))  float        v8f;
typedef __attribute__((ext_vector_type(4)))  unsigned int uv4;
typedef int v4i __attribute__((vector_size(16)));

union FragBF { uv4 q[2]; v16bf v; };

// -------- optional CDNA5 async global->LDS path (ASYNCcnt tracked) ---------
#if defined(__HIP_DEVICE_COMPILE__) && defined(__has_builtin)
#if __has_builtin(__builtin_amdgcn_global_load_async_to_lds_b128)
#define HAVE_ASYNC_LDS 1
#endif
#endif

#if defined(HAVE_ASYNC_LDS)
typedef __attribute__((address_space(1))) v4i* g_v4i_ptr;
typedef __attribute__((address_space(3))) v4i* l_v4i_ptr;
__device__ __forceinline__ void async_g2l_b128(const __bf16* g, __bf16* l) {
  __builtin_amdgcn_global_load_async_to_lds_b128(
      (g_v4i_ptr)(g), (l_v4i_ptr)(l), 0, 0);
}
__device__ __forceinline__ void wait_async0() {
#if __has_builtin(__builtin_amdgcn_s_wait_asynccnt)
  __builtin_amdgcn_s_wait_asynccnt(0);
#else
  asm volatile("s_wait_asynccnt 0x0" ::: "memory");
#endif
}
#endif

// A-matrix fragment (16x32 bf16), per ISA 7.12.2 16-bit A layout:
// lane L<16: row=L,  K = {kb..kb+7, kb+16..kb+23} with kb=0
// lane L>=16: row=L-16, same K pattern with kb=8
__device__ __forceinline__ v16bf frag_a(const __bf16* base, int pitch,
                                        int rowoff, int koff, int lane) {
  const int r  = rowoff + (lane & 15);
  const int kb = koff + ((lane >> 4) << 3);
  const char* p = (const char*)(base + (size_t)r * pitch + kb);
  FragBF f;
  f.q[0] = *(const uv4*)(p);
  f.q[1] = *(const uv4*)(p + 32);   // k + 16 elements
  return f.v;
}

// B-matrix fragment (32x16 bf16) from an N-major (transposed) buffer:
// lane L holds column n = L&15; K = {kb..kb+7, kb+16..kb+23} (kb = 0 or 8).
__device__ __forceinline__ v16bf frag_b(const __bf16* base, int pitch,
                                        int coloff, int koff, int lane) {
  const int n  = coloff + (lane & 15);
  const int kb = koff + ((lane >> 4) << 3);
  const char* p = (const char*)(base + (size_t)n * pitch + kb);
  FragBF f;
  f.q[0] = *(const uv4*)(p);
  f.q[1] = *(const uv4*)(p + 32);
  return f.v;
}

__device__ __forceinline__ v8f wmma_bf16(v16bf a, v16bf b, v8f c) {
  return __builtin_amdgcn_wmma_f32_16x16x32_bf16(false, a, false, b,
                                                 (short)0, c, false, false);
}

__device__ __forceinline__ float swishf(float x) {
  return x / (1.0f + __expf(-x));
}

// ---------------------------------------------------------------------------
// Weight prep: f32 (K x Nsrc) row-major  ->  bf16 transposed (Npad x Kpad)
// ---------------------------------------------------------------------------
__global__ void convert_wt(const float* __restrict__ W, int Ksrc, int Nsrc,
                           __bf16* __restrict__ Bt, int Kpad, int Npad) {
  int idx = blockIdx.x * blockDim.x + threadIdx.x;
  if (idx >= Kpad * Npad) return;
  int n = idx / Kpad, k = idx % Kpad;
  float v = (n < Nsrc && k < Ksrc) ? W[(size_t)k * Nsrc + n] : 0.0f;
  Bt[idx] = (__bf16)v;
}

__global__ void pad_bias(const float* __restrict__ b, int n,
                         float* __restrict__ dst, int npad) {
  int i = blockIdx.x * blockDim.x + threadIdx.x;
  if (i < npad) dst[i] = (i < n) ? b[i] : 0.0f;
}

// ---------------------------------------------------------------------------
// h = emb[z]   (N x 32, padded f32)
// ---------------------------------------------------------------------------
__global__ void node_emb(const int* __restrict__ z, const float* __restrict__ emb,
                         float* __restrict__ h, int N) {
  int n = blockIdx.x * blockDim.x + threadIdx.x;
  if (n >= N) return;
  int zi = z[n];
  #pragma unroll
  for (int u = 0; u < MULC; ++u) h[n * 32 + u] = emb[zi * MULC + u];
  h[n * 32 + 30] = 0.0f;
  h[n * 32 + 31] = 0.0f;
}

// ---------------------------------------------------------------------------
// Per-edge: sh1 vector and Gaussian radial basis (bf16, padded 40 -> 64)
// ---------------------------------------------------------------------------
__global__ void edge_prep(const float* __restrict__ pos,
                          const int* __restrict__ edge_index, int E,
                          float* __restrict__ sh1, __bf16* __restrict__ X0) {
  int e = blockIdx.x * blockDim.x + threadIdx.x;
  if (e >= E) return;
  int s = edge_index[e], t = edge_index[E + e];
  float dx = pos[s * 3 + 0] - pos[t * 3 + 0];
  float dy = pos[s * 3 + 1] - pos[t * 3 + 1];
  float dz = pos[s * 3 + 2] - pos[t * 3 + 2];
  float d  = sqrtf(dx * dx + dy * dy + dz * dz);
  // sqrt(3)/d/sqrt(20)
  float sc = 1.7320508075688772f * 0.22360679774997896f / d;
  sh1[e * 4 + 0] = dx * sc;
  sh1[e * 4 + 1] = dy * sc;
  sh1[e * 4 + 2] = dz * sc;
  sh1[e * 4 + 3] = 0.0f;
  const float inv_w = 1.0f / 0.25f;          // width = CUTOFF/NBASIS
  const float cstep = 10.0f / 39.0f;         // linspace(0,10,40)
  size_t base = (size_t)e * K0P;
  #pragma unroll
  for (int j = 0; j < NBASIS; ++j) {
    float tt = (d - cstep * (float)j) * inv_w;
    X0[base + j] = (__bf16)__expf(-tt * tt);
  }
  #pragma unroll
  for (int j = NBASIS; j < K0P; ++j) X0[base + j] = (__bf16)0.0f;
}

// ---------------------------------------------------------------------------
// GEMM:  C(M x Npad) = act(A(M x K) @ Bt^T + bias), bf16 in/out, f32 accum.
// Block 128x128 tile, 8 waves (2M x 4N), wave tile 64x32 (4x2 WMMA tiles).
// Double-buffered LDS staging; async G->LDS when available.
// ---------------------------------------------------------------------------
template <int ACT>
__global__ __launch_bounds__(256)
void gemm_bias_act(const __bf16* __restrict__ A, int lda,
                   const __bf16* __restrict__ Bt, int ldb,
                   const float* __restrict__ bias,
                   __bf16* __restrict__ Cmat, int ldc, int M, int K) {
  __shared__ __bf16 As[2][128][40];   // 32 valid + 8 pad (bank spread)
  __shared__ __bf16 Bs[2][128][40];

  const int tid  = threadIdx.x;
  const int lane = tid & 31;
  const int wave = tid >> 5;
  const int wm   = wave & 1;   // M sub-block  (x64)
  const int wn   = wave >> 1;  // N sub-block  (x32)
  const int blockM = blockIdx.x * 128;
  const int blockN = blockIdx.y * 128;

  v8f acc[4][2];
  #pragma unroll
  for (int i = 0; i < 4; ++i)
    #pragma unroll
    for (int j = 0; j < 2; ++j)
      acc[i][j] = v8f{0.f, 0.f, 0.f, 0.f, 0.f, 0.f, 0.f, 0.f};

  const int srow = tid >> 1;
  const int scol = (tid & 1) * 16;
  const int garow = blockM + srow;
  const bool aval = garow < M;
  const __bf16* Ap = A + (size_t)garow * lda + scol;
  const __bf16* Bp = Bt + (size_t)(blockN + srow) * ldb + scol;
  const int nsteps = K >> 5;
  const uv4 zz = {0u, 0u, 0u, 0u};

#if defined(HAVE_ASYNC_LDS)
  if (!aval) {   // boundary rows: keep zeros in both A buffers
    *(uv4*)&As[0][srow][scol] = zz; *(uv4*)&As[0][srow][scol + 8] = zz;
    *(uv4*)&As[1][srow][scol] = zz; *(uv4*)&As[1][srow][scol + 8] = zz;
  }
  if (aval) {
    async_g2l_b128(Ap + 0, &As[0][srow][scol]);
    async_g2l_b128(Ap + 8, &As[0][srow][scol + 8]);
  }
  async_g2l_b128(Bp + 0, &Bs[0][srow][scol]);
  async_g2l_b128(Bp + 8, &Bs[0][srow][scol + 8]);

  for (int i = 0; i < nsteps; ++i) {
    const int buf = i & 1;
    wait_async0();
    __syncthreads();
    if (i + 1 < nsteps) {           // prefetch next tile into other buffer
      const int k0 = (i + 1) << 5;
      if (aval) {
        async_g2l_b128(Ap + k0,     &As[buf ^ 1][srow][scol]);
        async_g2l_b128(Ap + k0 + 8, &As[buf ^ 1][srow][scol + 8]);
      }
      async_g2l_b128(Bp + k0,     &Bs[buf ^ 1][srow][scol]);
      async_g2l_b128(Bp + k0 + 8, &Bs[buf ^ 1][srow][scol + 8]);
    }
    const __bf16* asb = &As[buf][0][0];
    const __bf16* bsb = &Bs[buf][0][0];
    v16bf bf0 = frag_b(bsb, 40, wn * 32 + 0,  0, lane);
    v16bf bf1 = frag_b(bsb, 40, wn * 32 + 16, 0, lane);
    #pragma unroll
    for (int mt = 0; mt < 4; ++mt) {
      v16bf af = frag_a(asb, 40, wm * 64 + mt * 16, 0, lane);
      acc[mt][0] = wmma_bf16(af, bf0, acc[mt][0]);
      acc[mt][1] = wmma_bf16(af, bf1, acc[mt][1]);
    }
  }
#else
  uv4 a0 = zz, a1 = zz, b0, b1;
  if (aval) {
    const uv4* gp = (const uv4*)Ap;
    a0 = gp[0]; a1 = gp[1];
  }
  { const uv4* gb = (const uv4*)Bp; b0 = gb[0]; b1 = gb[1]; }

  for (int i = 0; i < nsteps; ++i) {
    const int buf = i & 1;
    *(uv4*)&As[buf][srow][scol]     = a0;
    *(uv4*)&As[buf][srow][scol + 8] = a1;
    *(uv4*)&Bs[buf][srow][scol]     = b0;
    *(uv4*)&Bs[buf][srow][scol + 8] = b1;
    __syncthreads();
    if (i + 1 < nsteps) {           // prefetch next tile into registers
      const int k0 = (i + 1) << 5;
      if (aval) {
        const uv4* gp = (const uv4*)(Ap + k0);
        a0 = gp[0]; a1 = gp[1];
      }
      const uv4* gb = (const uv4*)(Bp + k0);
      b0 = gb[0]; b1 = gb[1];
    }
    const __bf16* asb = &As[buf][0][0];
    const __bf16* bsb = &Bs[buf][0][0];
    v16bf bf0 = frag_b(bsb, 40, wn * 32 + 0,  0, lane);
    v16bf bf1 = frag_b(bsb, 40, wn * 32 + 16, 0, lane);
    #pragma unroll
    for (int mt = 0; mt < 4; ++mt) {
      v16bf af = frag_a(asb, 40, wm * 64 + mt * 16, 0, lane);
      acc[mt][0] = wmma_bf16(af, bf0, acc[mt][0]);
      acc[mt][1] = wmma_bf16(af, bf1, acc[mt][1]);
    }
  }
#endif

  // epilogue: bias (+ swish) -> bf16 store
  #pragma unroll
  for (int nt = 0; nt < 2; ++nt) {
    int col = blockN + wn * 32 + nt * 16 + (lane & 15);
    float bv = bias[col];
    #pragma unroll
    for (int mt = 0; mt < 4; ++mt) {
      #pragma unroll
      for (int r = 0; r < 8; ++r) {
        int row = blockM + wm * 64 + mt * 16 + ((lane < 16) ? r : r + 8);
        if (row < M) {
          float v = acc[mt][nt][r] + bv;
          if (ACT) v = swishf(v);
          Cmat[(size_t)row * ldc + col] = (__bf16)v;
        }
      }
    }
  }
}

// ---------------------------------------------------------------------------
// Fused: w = X3 @ W3 + b3 (500->2700, never stored), contracted with
// xj = h[src] into m0 (60) / m1 (30) per edge (LDS accum), scattered with
// the sh0/sh1 factors into agg0/agg1 via f32 atomics.
// 128 edges per block; each wave owns 16 edges, A strip (16x512) in VGPRs.
// ---------------------------------------------------------------------------
__global__ __launch_bounds__(256)
void final1_einsum(const __bf16* __restrict__ X,        // E x 512
                   const __bf16* __restrict__ Bt,       // 2704 x 512 (N-major)
                   const float*  __restrict__ bias,     // 2704 (padded)
                   const float*  __restrict__ h,        // N x 32
                   const int*    __restrict__ edge_index,
                   const float*  __restrict__ sh1,      // E x 4
                   float* __restrict__ agg0,            // N x 64
                   float* __restrict__ agg1,            // N x 96
                   int E) {
  __shared__ __bf16 Bs[2][16][136];    // 16 N-rows x 128 K-chunk, double buf
  __shared__ float  macc[128][90];     // per-edge m0(60) | m1(30)
  __shared__ __bf16 xjs[128][32];      // xj = h[src] per edge
  __shared__ int    srcIdx[128];

  const int tid  = threadIdx.x;
  const int lane = tid & 31;
  const int wave = tid >> 5;
  const int e0   = blockIdx.x * 128;

  for (int i = tid; i < 128 * 90; i += 256) (&macc[0][0])[i] = 0.0f;
  if (tid < 128) {
    int e = e0 + tid;
    srcIdx[tid] = (e < E) ? edge_index[e] : 0;
  }
  __syncthreads();
  for (int i = tid; i < 128 * 32; i += 256) {
    int el = i >> 5, u = i & 31;
    xjs[el][u] = (u < MULC) ? (__bf16)h[srcIdx[el] * 32 + u] : (__bf16)0.0f;
  }

  // A strip (this wave's 16 edges x K=512) resident in VGPRs (16 fragments)
  v16bf areg[16];
  {
    const int arow = e0 + wave * 16 + (lane & 15);
    const int kbl  = ((lane >> 4) << 3);
    const uv4 zz = {0u, 0u, 0u, 0u};
    #pragma unroll
    for (int ks = 0; ks < 16; ++ks) {
      FragBF f;
      if (arow < E) {
        const char* p = (const char*)(X + (size_t)arow * HP + ks * 32 + kbl);
        f.q[0] = *(const uv4*)(p);
        f.q[1] = *(const uv4*)(p + 32);
      } else {
        f.q[0] = zz; f.q[1] = zz;
      }
      areg[ks] = f.v;
    }
  }

  const int rr = tid >> 4;             // staging: 16 rows x 128 cols, 16B/thd
  const int cc = (tid & 15) * 8;
  const int NCHUNK = NW1T * 4;

#if defined(HAVE_ASYNC_LDS)
  async_g2l_b128(Bt + (size_t)rr * HP + cc, &Bs[0][rr][cc]);
#else
  uv4 breg = *(const uv4*)(Bt + (size_t)rr * HP + cc);
#endif

  for (int jt = 0; jt < NW1T; ++jt) {
    v8f acc = {0.f, 0.f, 0.f, 0.f, 0.f, 0.f, 0.f, 0.f};
    #pragma unroll
    for (int kc = 0; kc < 4; ++kc) {
      const int ch  = jt * 4 + kc;
      const int buf = ch & 1;
#if defined(HAVE_ASYNC_LDS)
      wait_async0();
      __syncthreads();
      if (ch + 1 < NCHUNK) {
        const int njt = (ch + 1) >> 2, nkc = (ch + 1) & 3;
        async_g2l_b128(Bt + (size_t)(njt * 16 + rr) * HP + nkc * 128 + cc,
                       &Bs[buf ^ 1][rr][cc]);
      }
#else
      *(uv4*)&Bs[buf][rr][cc] = breg;
      __syncthreads();
      if (ch + 1 < NCHUNK) {
        const int njt = (ch + 1) >> 2, nkc = (ch + 1) & 3;
        breg = *(const uv4*)(Bt + (size_t)(njt * 16 + rr) * HP + nkc * 128 + cc);
      }
#endif
      #pragma unroll
      for (int ks = 0; ks < 4; ++ks)
        acc = wmma_bf16(areg[kc * 4 + ks],
                        frag_b(&Bs[buf][0][0], 136, 0, ks * 32, lane), acc);
    }
    // contract this 16x16 w-tile with xj = h[src]
    const int nb = jt * 16;
    int col = nb + (lane & 15);
    float bv = bias[col];
    if (col < NW1) {
      int u, v;
      if (col < 1800) { u = col / 60; v = col % 60; }
      else            { int j = col - 1800; u = j / 30; v = 60 + j % 30; }
      #pragma unroll
      for (int r = 0; r < 8; ++r) {
        int rl = wave * 16 + ((lane < 16) ? r : r + 8);
        float xj = (float)xjs[rl][u];
        macc[rl][v] += (acc[r] + bv) * xj;   // addresses provably unique/lane
      }
    }
  }
  __syncthreads();

  // scatter to nodes
  const float inv_sqrt_mul = 0.18257418583505536f;                  // 30^-0.5
  const float m0scale = 0.22360679774997896f * inv_sqrt_mul;        // sh0/sqrt30
  for (int i = tid; i < 128 * 90; i += 256) {
    int el = i / 90, jj = i % 90;
    int e = e0 + el;
    if (e >= E) continue;
    int dn = edge_index[E + e];
    float val = macc[el][jj];
    if (jj < 60) {
      atomicAdd(&agg0[dn * 64 + jj], val * m0scale);
    } else {
      int u = jj - 60;
      float b = val * inv_sqrt_mul;
      atomicAdd(&agg1[dn * 96 + u * 3 + 0], b * sh1[e * 4 + 0]);
      atomicAdd(&agg1[dn * 96 + u * 3 + 1], b * sh1[e * 4 + 1]);
      atomicAdd(&agg1[dn * 96 + u * 3 + 2], b * sh1[e * 4 + 2]);
    }
  }
}

// ---------------------------------------------------------------------------
// Node update: sc = C*(h@W_si1)/sqrt30 + C*agg0 ; hs=swish ; g=sigmoid ;
// hv = g*agg1 ; out[batch] += C * (hs@W_si2)/sqrt30
// ---------------------------------------------------------------------------
__global__ __launch_bounds__(256)
void node_update(const float* __restrict__ h, const float* __restrict__ W_si1,
                 const float* __restrict__ W_si2, const float* __restrict__ agg0,
                 const float* __restrict__ agg1, const int* __restrict__ batch,
                 float* __restrict__ hs, float* __restrict__ hv,
                 float* __restrict__ out, int N) {
  __shared__ float outAcc[NGRAPH];
  int tid = threadIdx.x;
  if (tid < NGRAPH) outAcc[tid] = 0.0f;
  __syncthreads();
  int n = blockIdx.x * blockDim.x + tid;
  if (n < N) {
    const float Cc  = 0.70710678118654752f;
    const float ism = 0.18257418583505536f;
    float hh[MULC], hsv[MULC];
    #pragma unroll
    for (int u = 0; u < MULC; ++u) hh[u] = h[n * 32 + u];
    #pragma unroll
    for (int j = 0; j < MULC; ++j) {
      float s1 = 0.0f;
      #pragma unroll
      for (int u = 0; u < MULC; ++u) s1 += hh[u] * W_si1[u * 60 + j];
      float scv = Cc * (s1 * ism) + Cc * agg0[n * 64 + j];
      float v = swishf(scv);
      hsv[j] = v;
      hs[n * 32 + j] = v;
    }
    #pragma unroll
    for (int j = 0; j < MULC; ++j) {
      float s1 = 0.0f;
      #pragma unroll
      for (int u = 0; u < MULC; ++u) s1 += hh[u] * W_si1[u * 60 + 30 + j];
      float scv = Cc * (s1 * ism) + Cc * agg0[n * 64 + 30 + j];
      float g = 1.0f / (1.0f + __expf(-scv));
      hv[n * 96 + j * 3 + 0] = g * agg1[n * 96 + j * 3 + 0];
      hv[n * 96 + j * 3 + 1] = g * agg1[n * 96 + j * 3 + 1];
      hv[n * 96 + j * 3 + 2] = g * agg1[n * 96 + j * 3 + 2];
    }
    float s2 = 0.0f;
    #pragma unroll
    for (int u = 0; u < MULC; ++u) s2 += hsv[u] * W_si2[u];
    atomicAdd(&outAcc[batch[n]], Cc * (s2 * ism));
  }
  __syncthreads();
  if (tid < NGRAPH) atomicAdd(&out[tid], outAcc[tid]);
}

// ---------------------------------------------------------------------------
// Edge message 2: m = (w20.hs_src)*sh0 + (w21.dotv)/sqrt3, /sqrt60, *C,
// scattered to out[batch[dst]] with LDS block reduction.
// ---------------------------------------------------------------------------
__global__ __launch_bounds__(256)
void edge_msg2(const __bf16* __restrict__ W2,   // E x 128 (60 valid)
               const float* __restrict__ hs, const float* __restrict__ hv,
               const float* __restrict__ sh1, const int* __restrict__ edge_index,
               const int* __restrict__ batch, float* __restrict__ out, int E) {
  __shared__ float outAcc[NGRAPH];
  int tid = threadIdx.x;
  if (tid < NGRAPH) outAcc[tid] = 0.0f;
  __syncthreads();
  int e = blockIdx.x * blockDim.x + tid;
  if (e < E) {
    int s  = edge_index[e];
    int dn = edge_index[E + e];
    float shx = sh1[e * 4 + 0], shy = sh1[e * 4 + 1], shz = sh1[e * 4 + 2];
    float s0 = 0.0f, s1 = 0.0f;
    #pragma unroll
    for (int u = 0; u < MULC; ++u) {
      float w20 = (float)W2[(size_t)e * NW2P + u];
      float w21 = (float)W2[(size_t)e * NW2P + 30 + u];
      s0 += w20 * hs[s * 32 + u];
      float dot = hv[s * 96 + u * 3 + 0] * shx +
                  hv[s * 96 + u * 3 + 1] * shy +
                  hv[s * 96 + u * 3 + 2] * shz;
      s1 += w21 * dot;
    }
    float m = (s0 * 0.22360679774997896f + s1 * 0.57735026918962576f) *
              0.12909944487358056f;                       // * (2*MUL)^-0.5
    atomicAdd(&outAcc[batch[dn]], 0.70710678118654752f * m);
  }
  __syncthreads();
  if (tid < NGRAPH) atomicAdd(&out[tid], outAcc[tid]);
}

// ---------------------------------------------------------------------------
extern "C" void kernel_launch(void* const* d_in, const int* in_sizes, int n_in,
                              void* d_out, int out_size, void* d_ws, size_t ws_size,
                              hipStream_t stream) {
  const int*   z          = (const int*)d_in[0];
  const float* pos        = (const float*)d_in[1];
  const int*   batch      = (const int*)d_in[2];
  const int*   edge_index = (const int*)d_in[3];
  const float* emb        = (const float*)d_in[4];
  const float* W_si1      = (const float*)d_in[5];
  const float* W_si2      = (const float*)d_in[6];
  const float* r1w[4] = {(const float*)d_in[7],  (const float*)d_in[9],
                         (const float*)d_in[11], (const float*)d_in[13]};
  const float* r1b[4] = {(const float*)d_in[8],  (const float*)d_in[10],
                         (const float*)d_in[12], (const float*)d_in[14]};
  const float* r2w[4] = {(const float*)d_in[15], (const float*)d_in[17],
                         (const float*)d_in[19], (const float*)d_in[21]};
  const float* r2b[4] = {(const float*)d_in[16], (const float*)d_in[18],
                         (const float*)d_in[20], (const float*)d_in[22]};
  float* out = (float*)d_out;

  const int N = in_sizes[0];
  const int E = in_sizes[3] / 2;

  char* ws = (char*)d_ws;
  size_t off = 0;
  auto take = [&](size_t bytes) -> char* {
    char* p = ws + off;
    off = (off + bytes + 255) & ~(size_t)255;
    return p;
  };
  float*  h    = (float*)take((size_t)N * 32 * 4);
  float*  hsb  = (float*)take((size_t)N * 32 * 4);
  float*  hvb  = (float*)take((size_t)N * 96 * 4);
  float*  agg0 = (float*)take((size_t)N * 64 * 4);
  float*  agg1 = (float*)take((size_t)N * 96 * 4);
  float*  sh1  = (float*)take((size_t)E * 4 * 4);
  __bf16* X0   = (__bf16*)take((size_t)E * K0P * 2);
  __bf16* Xa   = (__bf16*)take((size_t)E * HP * 2);
  __bf16* Xb   = (__bf16*)take((size_t)E * HP * 2);
  __bf16* W2o  = (__bf16*)take((size_t)E * NW2P * 2);
  __bf16* Bt10 = (__bf16*)take((size_t)HP * K0P * 2);
  __bf16* Bt11 = (__bf16*)take((size_t)HP * HP * 2);
  __bf16* Bt12 = (__bf16*)take((size_t)HP * HP * 2);
  __bf16* Bt13 = (__bf16*)take((size_t)NW1P * HP * 2);
  __bf16* Bt20 = (__bf16*)take((size_t)HP * K0P * 2);
  __bf16* Bt21 = (__bf16*)take((size_t)HP * HP * 2);
  __bf16* Bt22 = (__bf16*)take((size_t)HP * HP * 2);
  __bf16* Bt23 = (__bf16*)take((size_t)NW2P * HP * 2);
  float* pb10 = (float*)take(HP * 4);
  float* pb11 = (float*)take(HP * 4);
  float* pb12 = (float*)take(HP * 4);
  float* pb13 = (float*)take(NW1P * 4);
  float* pb20 = (float*)take(HP * 4);
  float* pb21 = (float*)take(HP * 4);
  float* pb22 = (float*)take(HP * 4);
  float* pb23 = (float*)take(NW2P * 4);

  (void)hipMemsetAsync(out, 0, NGRAPH * sizeof(float), stream);
  (void)hipMemsetAsync(agg0, 0, (size_t)N * 64 * 4, stream);
  (void)hipMemsetAsync(agg1, 0, (size_t)N * 96 * 4, stream);

  auto cvt = [&](const float* W, int Ks, int Ns, __bf16* Bt, int Kp, int Np) {
    int tot = Kp * Np;
    convert_wt<<<(tot + 255) / 256, 256, 0, stream>>>(W, Ks, Ns, Bt, Kp, Np);
  };
  cvt(r1w[0], NBASIS, HDIM, Bt10, K0P, HP);
  cvt(r1w[1], HDIM, HDIM, Bt11, HP, HP);
  cvt(r1w[2], HDIM, HDIM, Bt12, HP, HP);
  cvt(r1w[3], HDIM, NW1,  Bt13, HP, NW1P);
  cvt(r2w[0], NBASIS, HDIM, Bt20, K0P, HP);
  cvt(r2w[1], HDIM, HDIM, Bt21, HP, HP);
  cvt(r2w[2], HDIM, HDIM, Bt22, HP, HP);
  cvt(r2w[3], HDIM, NW2,  Bt23, HP, NW2P);

  auto pb = [&](const float* b, int n, float* d, int np) {
    pad_bias<<<(np + 255) / 256, 256, 0, stream>>>(b, n, d, np);
  };
  pb(r1b[0], HDIM, pb10, HP);
  pb(r1b[1], HDIM, pb11, HP);
  pb(r1b[2], HDIM, pb12, HP);
  pb(r1b[3], NW1,  pb13, NW1P);
  pb(r2b[0], HDIM, pb20, HP);
  pb(r2b[1], HDIM, pb21, HP);
  pb(r2b[2], HDIM, pb22, HP);
  pb(r2b[3], NW2,  pb23, NW2P);

  node_emb<<<(N + 255) / 256, 256, 0, stream>>>(z, emb, h, N);
  edge_prep<<<(E + 255) / 256, 256, 0, stream>>>(pos, edge_index, E, sh1, X0);

  const int mg = (E + 127) / 128;

  // radial MLP 1 (WMMA GEMMs), last layer fused below
  gemm_bias_act<1><<<dim3(mg, HP / 128), 256, 0, stream>>>(X0, K0P, Bt10, K0P, pb10, Xa, HP, E, K0P);
  gemm_bias_act<1><<<dim3(mg, HP / 128), 256, 0, stream>>>(Xa, HP,  Bt11, HP,  pb11, Xb, HP, E, HP);
  gemm_bias_act<1><<<dim3(mg, HP / 128), 256, 0, stream>>>(Xb, HP,  Bt12, HP,  pb12, Xa, HP, E, HP);

  final1_einsum<<<mg, 256, 0, stream>>>(Xa, Bt13, pb13, h, edge_index, sh1,
                                        agg0, agg1, E);

  node_update<<<(N + 255) / 256, 256, 0, stream>>>(h, W_si1, W_si2, agg0, agg1,
                                                   batch, hsb, hvb, out, N);

  // radial MLP 2 (WMMA GEMMs), final layer materialized (only 25 MB)
  gemm_bias_act<1><<<dim3(mg, HP / 128), 256, 0, stream>>>(X0, K0P, Bt20, K0P, pb20, Xb, HP, E, K0P);
  gemm_bias_act<1><<<dim3(mg, HP / 128), 256, 0, stream>>>(Xb, HP,  Bt21, HP,  pb21, Xa, HP, E, HP);
  gemm_bias_act<1><<<dim3(mg, HP / 128), 256, 0, stream>>>(Xa, HP,  Bt22, HP,  pb22, Xb, HP, E, HP);
  gemm_bias_act<0><<<dim3(mg, NW2P / 128), 256, 0, stream>>>(Xb, HP, Bt23, HP, pb23, W2o, NW2P, E, HP);

  edge_msg2<<<(E + 255) / 256, 256, 0, stream>>>(W2o, hsb, hvb, sh1, edge_index,
                                                 batch, out, E);
}
